// NonLocalAttention_18133351923962
// MI455X (gfx1250) — matrix operand order repaired and checked by reading, not compiled, not run on hardware
//
#include <hip/hip_runtime.h>
#include <hip/hip_bf16.h>

// Problem constants (match reference: N=2, C=128, H=W=96, Cr=C/2)
constexpr int NB = 2;
constexpr int Cc = 128;
constexpr int CR = 64;
constexpr int LL = 96 * 96;   // 9216
constexpr float LOG2E = 1.44269504088896340736f;

typedef __bf16 bf16x16 __attribute__((ext_vector_type(16)));
typedef float  f32x8   __attribute__((ext_vector_type(8)));

union FragBF {
    bf16x16 v;
    unsigned int u[8];
};
union FragF {
    f32x8 v;
    float f[8];
};

// Cheap f32 -> bf16 (round-half-up): add + shift.
__device__ __forceinline__ unsigned short f2bf(float x) {
    union { float f; unsigned int u; } a; a.f = x;
    return (unsigned short)((a.u + 0x8000u) >> 16);
}
// Pack two f32 into packed bf16x2: 2x v_add + 1x v_perm_b32.
__device__ __forceinline__ unsigned int pack2(float lo, float hi) {
    union { float f; unsigned int u; } A, B; A.f = lo; B.f = hi;
    const unsigned int al = A.u + 0x8000u;
    const unsigned int bh = B.u + 0x8000u;
    return __builtin_amdgcn_perm(bh, al, 0x07060302u);  // {bh[3],bh[2],al[3],al[2]}
}

// gfx1250 async global->LDS copy path (ASYNCcnt-tracked, no VGPR round trip).
// Builtin signature (from hipcc diagnostic): param0 = int4 AS(1)*, i.e. a
// global-address-space vector pointer; LDS side is AS(3).
#if __has_builtin(__builtin_amdgcn_global_load_async_to_lds_b128) && \
    __has_builtin(__builtin_amdgcn_s_wait_asynccnt)
#define USE_ASYNC_LDS 1
typedef int v4i __attribute__((vector_size(16)));
typedef v4i __attribute__((address_space(1))) g4i;   // global int4
typedef v4i __attribute__((address_space(3))) l4i;   // LDS int4
#else
#define USE_ASYNC_LDS 0
#endif

// ---------------------------------------------------------------------------
// Prep 1: pack the three weight matrices to bf16 (u32-pair granularity).
// ---------------------------------------------------------------------------
__global__ __launch_bounds__(256) void pack_weights(
    const float* __restrict__ w1, const float* __restrict__ w2,
    const float* __restrict__ wa,
    unsigned short* __restrict__ o1, unsigned short* __restrict__ o2,
    unsigned short* __restrict__ oa)
{
    const int i = blockIdx.x * 256 + threadIdx.x;     // u32 index
    const int n1 = CR * Cc / 2;                       // 4096
    const int na = Cc * Cc / 2;                       // 8192
    if (i < n1) {
        ((unsigned int*)o1)[i] = pack2(w1[2 * i], w1[2 * i + 1]);
    } else if (i < 2 * n1) {
        const int j = i - n1;
        ((unsigned int*)o2)[j] = pack2(w2[2 * j], w2[2 * j + 1]);
    } else if (i < 2 * n1 + na) {
        const int j = i - 2 * n1;
        ((unsigned int*)oa)[j] = pack2(wa[2 * j], wa[2 * j + 1]);
    }
}

// ---------------------------------------------------------------------------
// Prep 2: x [N][C][L] f32  ->  xT [N][L][C] bf16 (A-fragment friendly).
// ---------------------------------------------------------------------------
__global__ __launch_bounds__(256) void xpose(
    const float* __restrict__ x, unsigned short* __restrict__ xT)
{
    const int n   = blockIdx.y;
    const int row = blockIdx.x * 256 + threadIdx.x;   // l index
    const float* xr = x + ((size_t)n * Cc) * LL + row;
    unsigned int* o = (unsigned int*)(xT + ((size_t)n * LL + row) * Cc);
#pragma unroll
    for (int c = 0; c < Cc; c += 2)
        o[c >> 1] = pack2(xr[(size_t)c * LL], xr[(size_t)(c + 1) * LL]);
}

// ---------------------------------------------------------------------------
// Phase 1: 1x1 conv + PReLU as pure-bf16 WMMA GEMM.
//   y[l][co] = scaleOut * prelu(sum_c xT[l][c]*w[co][c] + b[co])
// transposeOut=0 -> out [n][l][Co]   (q, k)
// transposeOut=1 -> out [n][co][L]   (v), stored as packed uint4 row-pairs
// ---------------------------------------------------------------------------
__global__ __launch_bounds__(128) void qkv_gemm(
    const unsigned short* __restrict__ xT,  // [N][L][C] bf16
    const unsigned short* __restrict__ wb,  // [Co][C]   bf16
    const float* __restrict__ b,            // [Co] f32
    const float* __restrict__ al,           // scalar
    float scaleOut,
    unsigned short* __restrict__ out,
    int Co, int transposeOut)
{
    const int n    = blockIdx.y;
    const int wave = threadIdx.x >> 5;
    const int lane = threadIdx.x & 31;
    const int half = lane >> 4;
    const int l15  = lane & 15;
    const int lbase = blockIdx.x * 64 + wave * 16;
    const int row  = lbase + l15;
    const float a  = al[0];

    FragBF A[4];
    const unsigned int* xr = (const unsigned int*)(xT + ((size_t)n * LL + row) * Cc);
#pragma unroll
    for (int kk = 0; kk < 4; ++kk) {
        const int base = kk * 16 + half * 4;
#pragma unroll
        for (int j = 0; j < 4; ++j) {
            A[kk].u[j]     = xr[base + j];
            A[kk].u[j + 4] = xr[base + 8 + j];
        }
    }

    for (int cf = 0; cf < (Co >> 4); ++cf) {
        FragF acc;
#pragma unroll
        for (int i = 0; i < 8; ++i) acc.f[i] = 0.f;
        const int co = cf * 16 + l15;
        const unsigned int* wr = (const unsigned int*)(wb + (size_t)co * Cc);
#pragma unroll
        for (int kk = 0; kk < 4; ++kk) {
            FragBF B;
            const int base = kk * 16 + half * 4;
#pragma unroll
            for (int j = 0; j < 8; ++j) B.u[j] = wr[base + (j & 3) + (j >> 2) * 8];
            acc.v = __builtin_amdgcn_wmma_f32_16x16x32_bf16(
                        false, A[kk].v, false, B.v, (short)0, acc.v, false, false);
        }
        const float bias = b[co];
        float y[8];
#pragma unroll
        for (int r = 0; r < 8; ++r) {
            float t = acc.f[r] + bias;
            t = fmaxf(t, 0.f) + a * fminf(t, 0.f);    // PReLU
            y[r] = t * scaleOut;
        }
        if (transposeOut) {
            uint4 st;
            st.x = pack2(y[0], y[1]); st.y = pack2(y[2], y[3]);
            st.z = pack2(y[4], y[5]); st.w = pack2(y[6], y[7]);
            *(uint4*)(out + (size_t)(n * Co + co) * LL + lbase + half * 8) = st;
        } else {
#pragma unroll
            for (int r = 0; r < 8; ++r) {
                const int rr = lbase + r + half * 8;
                out[((size_t)n * LL + rr) * (size_t)Co + co] = f2bf(y[r]);
            }
        }
    }
}

// ---------------------------------------------------------------------------
// Phase 2: flash-attention fusion (base-2 softmax; q pre-scaled by log2e).
// Block = 128 threads = 4 waves; wave owns 16 query rows; key tiles of 64.
// LDS rows padded by 8 shorts: 16B-aligned, row stride = 36 dwords ->
// 16-lane row gathers are bank-conflict-free.
// ---------------------------------------------------------------------------
__global__ __launch_bounds__(128) void flash_attn(
    const unsigned short* __restrict__ qg,   // [N][L][Cr] bf16 (x log2e)
    const unsigned short* __restrict__ kg,   // [N][L][Cr] bf16
    const unsigned short* __restrict__ vgT,  // [N][C][L]  bf16
    const float* __restrict__ xres,          // [N][C][L]  f32 residual
    float* __restrict__ out)                 // [N][C][L]  f32
{
    __shared__ __attribute__((aligned(16))) unsigned short kSh[64][72];   // [key][ch]
    __shared__ __attribute__((aligned(16))) unsigned short vSh[128][72];  // [ch][key]
    __shared__ __attribute__((aligned(16))) unsigned short pSh[4][16][72];

    const int n    = blockIdx.y;
    const int wave = threadIdx.x >> 5;
    const int lane = threadIdx.x & 31;
    const int half = lane >> 4;
    const int l15  = lane & 15;
    const int tid  = threadIdx.x;
    const int lbase = blockIdx.x * 64 + wave * 16;
    const int row  = lbase + l15;

    // q A-fragments: K = Cr = 64 -> 2 ksteps, contiguous u32 loads.
    FragBF qA[2];
    const unsigned int* qr = (const unsigned int*)(qg + ((size_t)n * LL + row) * CR);
#pragma unroll
    for (int kk = 0; kk < 2; ++kk) {
        const int base = kk * 16 + half * 4;
#pragma unroll
        for (int j = 0; j < 4; ++j) {
            qA[kk].u[j]     = qr[base + j];
            qA[kk].u[j + 4] = qr[base + 8 + j];
        }
    }

    FragF o[8];
#pragma unroll
    for (int of = 0; of < 8; ++of)
#pragma unroll
        for (int i = 0; i < 8; ++i) o[of].f[i] = 0.f;
    float m[8], lsum[8];
#pragma unroll
    for (int r = 0; r < 8; ++r) { m[r] = -3.0e38f; lsum[r] = 0.f; }

    for (int t = 0; t < LL / 64; ++t) {
        const int kb = t * 64;
        __syncthreads();   // previous tile fully consumed
        // Stage k tile (64 keys x 64 ch) and v tile (128 ch x 64 keys).
        {
            const int key = tid >> 1;
            const unsigned short* src  = kg  + ((size_t)(n * LL + kb + key)) * CR;
            const unsigned short* vsrc = vgT + ((size_t)(n * Cc + tid)) * LL + kb;
#if USE_ASYNC_LDS
#pragma unroll
            for (int q4 = 0; q4 < 4; ++q4) {
                const int qq = (tid & 1) * 4 + q4;
                __builtin_amdgcn_global_load_async_to_lds_b128(
                    (g4i*)(src + qq * 8), (l4i*)&kSh[key][qq * 8], 0, 0);
            }
#pragma unroll
            for (int q8 = 0; q8 < 8; ++q8)
                __builtin_amdgcn_global_load_async_to_lds_b128(
                    (g4i*)(vsrc + q8 * 8), (l4i*)&vSh[tid][q8 * 8], 0, 0);
            __builtin_amdgcn_s_wait_asynccnt(0);   // LDS writes landed
#else
#pragma unroll
            for (int q4 = 0; q4 < 4; ++q4) {
                const int qq = (tid & 1) * 4 + q4;
                *(uint4*)(&kSh[key][qq * 8]) = *(const uint4*)(src + qq * 8);
            }
#pragma unroll
            for (int q8 = 0; q8 < 8; ++q8)
                *(uint4*)(&vSh[tid][q8 * 8]) = *(const uint4*)(vsrc + q8 * 8);
#endif
        }
        __syncthreads();

        // S = q @ k^T  (16 x 64, f32 accum).
        FragF s[4];
#pragma unroll
        for (int nf = 0; nf < 4; ++nf) {
#pragma unroll
            for (int i = 0; i < 8; ++i) s[nf].f[i] = 0.f;
            const unsigned int* kr = (const unsigned int*)&kSh[nf * 16 + l15][0];
#pragma unroll
            for (int kk = 0; kk < 2; ++kk) {
                FragBF B;
                const int base = kk * 16 + half * 4;
#pragma unroll
                for (int j = 0; j < 8; ++j) B.u[j] = kr[base + (j & 3) + (j >> 2) * 8];
                s[nf].v = __builtin_amdgcn_wmma_f32_16x16x32_bf16(
                              false, qA[kk].v, false, B.v, (short)0, s[nf].v, false, false);
            }
        }

        // Online softmax in base 2 (xor-reduce within 16-lane halves).
#pragma unroll
        for (int r = 0; r < 8; ++r) {
            float lm = fmaxf(fmaxf(s[0].f[r], s[1].f[r]), fmaxf(s[2].f[r], s[3].f[r]));
#pragma unroll
            for (int mk = 1; mk < 16; mk <<= 1) lm = fmaxf(lm, __shfl_xor(lm, mk, 32));
            const float mn   = fmaxf(m[r], lm);
            const float alph = __builtin_amdgcn_exp2f(m[r] - mn);
            float rs = 0.f;
#pragma unroll
            for (int nf = 0; nf < 4; ++nf) {
                const float p = __builtin_amdgcn_exp2f(s[nf].f[r] - mn);
                s[nf].f[r] = p;
                rs += p;
            }
#pragma unroll
            for (int mk = 1; mk < 16; mk <<= 1) rs += __shfl_xor(rs, mk, 32);
            lsum[r] = lsum[r] * alph + rs;
            m[r] = mn;
#pragma unroll
            for (int of = 0; of < 8; ++of) o[of].f[r] *= alph;
        }

        // Re-layout P: C-fragment -> wave-private LDS -> A-fragment.
#pragma unroll
        for (int nf = 0; nf < 4; ++nf)
#pragma unroll
            for (int r = 0; r < 8; ++r)
                pSh[wave][r + half * 8][nf * 16 + l15] = f2bf(s[nf].f[r]);

        // O += P @ V  (16 x 128, K = 64 keys -> 2 ksteps).
#pragma unroll
        for (int kk = 0; kk < 2; ++kk) {
            FragBF A;
            const unsigned int* pr = (const unsigned int*)&pSh[wave][l15][0];
            const int abase = kk * 16 + half * 4;
#pragma unroll
            for (int j = 0; j < 4; ++j) {
                A.u[j]     = pr[abase + j];
                A.u[j + 4] = pr[abase + 8 + j];
            }
#pragma unroll
            for (int of = 0; of < 8; ++of) {
                FragBF B;
                const unsigned int* vr = (const unsigned int*)&vSh[of * 16 + l15][0];
                const int base = kk * 16 + half * 4;
#pragma unroll
                for (int j = 0; j < 8; ++j) B.u[j] = vr[base + (j & 3) + (j >> 2) * 8];
                o[of].v = __builtin_amdgcn_wmma_f32_16x16x32_bf16(
                              false, A.v, false, B.v, (short)0, o[of].v, false, false);
            }
        }
    }

    // Epilogue: normalize (v_rcp), add residual, store f32 [N][C][L].
    float inv[8];
#pragma unroll
    for (int r = 0; r < 8; ++r) inv[r] = __builtin_amdgcn_rcpf(lsum[r]);
#pragma unroll
    for (int of = 0; of < 8; ++of) {
        const int c = of * 16 + l15;
#pragma unroll
        for (int r = 0; r < 8; ++r) {
            const int rr = lbase + r + half * 8;
            const size_t idx = ((size_t)(n * Cc + c)) * LL + rr;
            out[idx] = o[of].f[r] * inv[r] + xres[idx];
        }
    }
}

extern "C" void kernel_launch(void* const* d_in, const int* in_sizes, int n_in,
                              void* d_out, int out_size, void* d_ws, size_t ws_size,
                              hipStream_t stream) {
    const float* x  = (const float*)d_in[0];
    const float* w1 = (const float*)d_in[1];
    const float* b1 = (const float*)d_in[2];
    const float* a1 = (const float*)d_in[3];
    const float* w2 = (const float*)d_in[4];
    const float* b2 = (const float*)d_in[5];
    const float* a2 = (const float*)d_in[6];
    const float* wa = (const float*)d_in[7];
    const float* ba = (const float*)d_in[8];
    const float* aa = (const float*)d_in[9];
    float* out = (float*)d_out;

    unsigned short* qg  = (unsigned short*)d_ws;         // [N][L][CR]
    unsigned short* kg  = qg  + (size_t)NB * LL * CR;    // [N][L][CR]
    unsigned short* vgT = kg  + (size_t)NB * LL * CR;    // [N][C][L]
    unsigned short* xT  = vgT + (size_t)NB * LL * Cc;    // [N][L][C]
    unsigned short* w1b = xT  + (size_t)NB * LL * Cc;    // [CR][C]
    unsigned short* w2b = w1b + (size_t)CR * Cc;
    unsigned short* wab = w2b + (size_t)CR * Cc;

    const int wU32 = (2 * CR * Cc + Cc * Cc) / 2;        // 16384 u32 elements
    pack_weights<<<(wU32 + 255) / 256, 256, 0, stream>>>(w1, w2, wa, w1b, w2b, wab);
    xpose<<<dim3(LL / 256, NB), 256, 0, stream>>>(x, xT);

    dim3 grid(LL / 64, NB);
    qkv_gemm<<<grid, 128, 0, stream>>>(xT, w1b, b1, a1, LOG2E, qg,  CR, 0);
    qkv_gemm<<<grid, 128, 0, stream>>>(xT, w2b, b2, a2, 1.0f,  kg,  CR, 0);
    qkv_gemm<<<grid, 128, 0, stream>>>(xT, wab, ba, aa, 1.0f,  vgT, Cc, 1);
    flash_attn<<<grid, 128, 0, stream>>>(qg, kg, vgT, x, out);
}
